// GraphEmbed_27582279975426
// MI455X (gfx1250) — matrix operand order repaired and checked by reading, not compiled, not run on hardware
//
#include <hip/hip_runtime.h>
#include <hip/hip_bf16.h>

// ---------------------------------------------------------------------------
// GraphEmbed pipeline for MI455X (gfx1250, wave32, WMMA).
//   deg/norm -> sigmoid-gating GEMM -> 2x TAGConv (SpMM hops + concat GEMM +
//   BatchNorm + ReLU) -> final GEMM fused with mean readout.
// Dense GEMMs: v_wmma_f32_16x16x32_f16. Each wave32 owns a 16x128 output
// strip (8 accumulators). Per K-step: load A once + all 8 B fragments into
// distinct registers (batched b128 clauses), then drain 8 WMMAs so loads and
// matrix ops overlap within the wave.
// Weights are pre-transposed to [Nc,K] so BOTH operands load as b128 vectors.
// ---------------------------------------------------------------------------

typedef __attribute__((ext_vector_type(16))) _Float16 v16h;
typedef __attribute__((ext_vector_type(8)))  _Float16 v8h;
typedef __attribute__((ext_vector_type(8)))  float    v8f;

#define DH    128      // hidden width
#define KCAT  384      // (K+1)*DH concat width
#define DOUT  256      // graph embedding size
#define BN_EPS 1e-5f

// ---------------------------- elementwise kernels --------------------------

__global__ void k_fill0(float* __restrict__ p, int n) {
  int i = blockIdx.x * blockDim.x + threadIdx.x;
  if (i < n) p[i] = 0.0f;
}

__global__ void k_deg(const int* __restrict__ dst, float* __restrict__ deg, int E) {
  int e = blockIdx.x * blockDim.x + threadIdx.x;
  if (e < E) atomicAdd(&deg[dst[e]], 1.0f);
}

__global__ void k_norm_inplace(float* __restrict__ d, int n) {
  int i = blockIdx.x * blockDim.x + threadIdx.x;
  if (i < n) {
    float v = d[i];
    v = v < 1.0f ? 1.0f : v;
    d[i] = rsqrtf(v);
  }
}

__global__ void k_cvt_f16(const float* __restrict__ s, _Float16* __restrict__ d, int n) {
  int i = blockIdx.x * blockDim.x + threadIdx.x;
  if (i < n) d[i] = (_Float16)s[i];
}

// transpose-convert W [K,Nc] f32 -> Wt [Nc,K] f16 (weights are small)
__global__ void k_cvt_f16_T(const float* __restrict__ s, _Float16* __restrict__ d,
                            int K, int Nc) {
  int i = blockIdx.x * blockDim.x + threadIdx.x;
  if (i < K * Nc) {
    int k = i / Nc, n = i - k * Nc;
    d[(long)n * K + k] = (_Float16)s[i];
  }
}

// copy h (f32, [N,DH]) into column block `off` of X (f16, [N,KCAT])
__global__ void k_copyX(const float* __restrict__ h, _Float16* __restrict__ X,
                        int n, int off) {
  int i = blockIdx.x * blockDim.x + threadIdx.x;
  if (i < n) {
    int r = i >> 7, c = i & 127;
    X[(long)r * KCAT + off + c] = (_Float16)h[i];
  }
}

// acc[dst[e]][c..c+3] += h[src[e]][c..c+3] * norm[src[e]]
// one thread per (edge, 4-column group): b128 load + 4 f32 atomics
__global__ void k_scatter(const int* __restrict__ src, const int* __restrict__ dst,
                          const float* __restrict__ norm, const float* __restrict__ h,
                          float* __restrict__ acc, int total) {
  int t = blockIdx.x * blockDim.x + threadIdx.x;
  if (t < total) {
    int e = t >> 5, q = (t & 31) << 2;
    int s = src[e], d = dst[e];
    float ns = norm[s];
    const float4 v = *(const float4*)(h + (long)s * DH + q);
    float* a = acc + (long)d * DH + q;
    atomicAdd(a + 0, v.x * ns);
    atomicAdd(a + 1, v.y * ns);
    atomicAdd(a + 2, v.z * ns);
    atomicAdd(a + 3, v.w * ns);
  }
}

// cur = acc * norm[row]; also write f16 copy into X column block `off`
__global__ void k_hopscale(const float* __restrict__ acc, const float* __restrict__ norm,
                           float* __restrict__ cur, _Float16* __restrict__ X,
                           int n, int off) {
  int i = blockIdx.x * blockDim.x + threadIdx.x;
  if (i < n) {
    int r = i >> 7, c = i & 127;
    float v = acc[i] * norm[r];
    cur[i] = v;
    X[(long)r * KCAT + off + c] = (_Float16)v;
  }
}

// per-block partial column sums / sums-of-squares over 256-row stripes
__global__ void k_bnstats(const float* __restrict__ x, float* __restrict__ sum,
                          float* __restrict__ sumsq, int rows) {
  int c = threadIdx.x;                 // 0..127
  int r0 = blockIdx.x * 256;
  int r1 = r0 + 256; if (r1 > rows) r1 = rows;
  float s = 0.0f, s2 = 0.0f;
  for (int r = r0; r < r1; ++r) {
    float v = x[(long)r * DH + c];
    s += v; s2 += v * v;
  }
  atomicAdd(&sum[c], s);
  atomicAdd(&sumsq[c], s2);
}

__global__ void k_bnfinal(const float* __restrict__ sum, const float* __restrict__ sumsq,
                          const float* __restrict__ g, const float* __restrict__ b,
                          float* __restrict__ scale, float* __restrict__ shift,
                          float invN) {
  int c = threadIdx.x;                 // 128 threads, 1 block
  float mu  = sum[c] * invN;
  float var = sumsq[c] * invN - mu * mu;
  float a   = g[c] * rsqrtf(var + BN_EPS);
  scale[c] = a;
  shift[c] = b[c] - mu * a;
}

__global__ void k_bnapply_relu(const float* __restrict__ x, const float* __restrict__ scale,
                               const float* __restrict__ shift, float* __restrict__ h,
                               int n) {
  int i = blockIdx.x * blockDim.x + threadIdx.x;
  if (i < n) {
    int c = i & 127;
    float v = x[i] * scale[c] + shift[c];
    h[i] = v > 0.0f ? v : 0.0f;
  }
}

__global__ void k_finalout(const float* __restrict__ colacc, const float* __restrict__ bd,
                           float* __restrict__ out, float invN, int n) {
  int c = blockIdx.x * blockDim.x + threadIdx.x;
  if (c < n) out[c] = colacc[c] * invN + bd[c];
}

// ------------------------------- WMMA GEMM ---------------------------------
// C[M,Nc] = A[M,K] (f16 row-major, lda) @ Bt[Nc,K] (f16 pre-transposed, ld=K).
// Each wave32 computes a 16x128 output strip. Per 32-wide K-step: one A
// fragment (2x b128) + eight B fragments (16x b128, distinct registers) are
// loaded, then 8 v_wmma_f32_16x16x32_f16 drain with pipelined loadcnt waits.
// Nc must be a multiple of 128; grid covers (M/16) * (Nc/128) waves.
// EPI: 0 = store f32, 1 = sigmoid(x + bias[col]) store, 2 = column-sum reduce
//      (atomicAdd into colsum[Nc], no C store) for the fused mean readout.
//
// VGPR layouts (CDNA5 ISA, 16-bit operands, wave32):
//   A 16x32: lane = half*16+m; holds K = half*8+{0..7} and 16+half*8+{0..7}
//            -> two contiguous 16B runs -> two b128 loads
//   B 32x16: lane = half*16+n; holds K = half*8+{0..7} and 16+half*8+{0..7}
//            -> two contiguous 16B runs in Bt row n -> two b128 loads
//   C/D 16x16 f32: VGPR j -> row = half*8 + j, col = lane&15
template <int EPI>
__global__ __launch_bounds__(256) void k_gemm_wmma(
    const _Float16* __restrict__ A, int lda,
    const _Float16* __restrict__ Bt,
    float* __restrict__ C, int ldc,
    int M, int K, int Nc,
    const float* __restrict__ bias, float* __restrict__ colsum) {
  constexpr int NT = 8;                    // column tiles per wave (128 cols)
  int lane = threadIdx.x & 31;
  int wave = blockIdx.x * (blockDim.x >> 5) + (threadIdx.x >> 5);
  int ngrp  = Nc >> 7;                     // 128-column groups
  int waves = (M >> 4) * ngrp;
  if (wave >= waves) return;               // wave-uniform: EXEC stays all-ones

  int tm = wave / ngrp;
  int tg = wave - tm * ngrp;
  int half  = lane >> 4;
  int idx16 = lane & 15;

  const _Float16* Ap = A  + (long)(tm * 16 + idx16) * lda + half * 8;
  const _Float16* Bp = Bt + ((long)tg * 128 + idx16) * K + half * 8;
  const long bstride = (long)16 * K;       // Bt rows per column tile

  v8f acc[NT] = {};

  for (int kb = 0; kb < K; kb += 32) {
    __builtin_prefetch(Ap + kb + 64, 0, 1);    // global_prefetch_b8
    // ---- load phase: A fragment + all 8 B fragments (distinct registers) ----
    v8h a0 = *(const v8h*)(Ap + kb);
    v8h a1 = *(const v8h*)(Ap + kb + 16);
    v16h bfr[NT];
#pragma unroll
    for (int t = 0; t < NT; ++t) {
      const _Float16* bp = Bp + t * bstride + kb;
      v8h b0 = *(const v8h*)(bp);
      v8h b1 = *(const v8h*)(bp + 16);
      bfr[t] = __builtin_shufflevector(b0, b1, 0,1,2,3,4,5,6,7,8,9,10,11,12,13,14,15);
    }
    v16h a = __builtin_shufflevector(a0, a1, 0,1,2,3,4,5,6,7,8,9,10,11,12,13,14,15);
    // ---- compute phase: drain 8 WMMAs against the shared A fragment ----
#pragma unroll
    for (int t = 0; t < NT; ++t) {
      acc[t] = __builtin_amdgcn_wmma_f32_16x16x32_f16(
          /*neg_a=*/false, a, /*neg_b=*/false, bfr[t],
          /*c_mod=*/(short)0, acc[t], /*reuse_a=*/false, /*reuse_b=*/false);
    }
  }

  if (EPI == 2) {                          // fused column-sum (mean readout)
#pragma unroll
    for (int t = 0; t < NT; ++t) {
      float s = 0.0f;
#pragma unroll
      for (int j = 0; j < 8; ++j) s += acc[t][j];
      atomicAdd(&colsum[tg * 128 + t * 16 + idx16], s);
    }
    return;
  }
#pragma unroll
  for (int t = 0; t < NT; ++t) {
    int n = tg * 128 + t * 16 + idx16;
#pragma unroll
    for (int j = 0; j < 8; ++j) {
      int row = tm * 16 + half * 8 + j;
      float v = acc[t][j];
      if (EPI == 1) v = 1.0f / (1.0f + __expf(-(v + bias[n])));
      C[(long)row * ldc + n] = v;
    }
  }
}

// ------------------------------- launcher ----------------------------------

static inline int ceil_div(long a, long b) { return (int)((a + b - 1) / b); }

extern "C" void kernel_launch(void* const* d_in, const int* in_sizes, int n_in,
                              void* d_out, int out_size, void* d_ws, size_t ws_size,
                              hipStream_t stream) {
  const float* features = (const float*)d_in[0];
  const int*   src      = (const int*)  d_in[1];
  const int*   dst      = (const int*)  d_in[2];
  const float* Wg       = (const float*)d_in[3];
  const float* bg       = (const float*)d_in[4];
  const float* Wc1      = (const float*)d_in[5];
  const float* g1       = (const float*)d_in[6];
  const float* b1       = (const float*)d_in[7];
  const float* Wc2      = (const float*)d_in[8];
  const float* g2       = (const float*)d_in[9];
  const float* b2       = (const float*)d_in[10];
  const float* Wd       = (const float*)d_in[11];
  const float* bd       = (const float*)d_in[12];
  float*       out      = (float*)d_out;

  const int N = in_sizes[0] / DH;          // 50000
  const int E = in_sizes[1];               // 800000
  const int NE = N * DH;                   // elements in an activation matrix
  const float invN = 1.0f / (float)N;

  // ---- carve workspace (~110 MB) ----
  size_t off = 0;
  auto carve = [&](size_t bytes) -> void* {
    void* p = (char*)d_ws + off;
    off += (bytes + 255) & ~(size_t)255;
    return p;
  };
  float*     norm   = (float*)    carve((size_t)N * 4);
  float*     bufH   = (float*)    carve((size_t)NE * 4);   // layer activations
  float*     bufAcc = (float*)    carve((size_t)NE * 4);   // SpMM accum / GEMM out
  float*     bufCur = (float*)    carve((size_t)NE * 4);   // current hop
  _Float16*  Xh     = (_Float16*) carve((size_t)N * KCAT * 2);  // concat (f16)
  _Float16*  WgT    = (_Float16*) carve((size_t)DH * DH * 2);
  _Float16*  Wc1T   = (_Float16*) carve((size_t)KCAT * DH * 2);
  _Float16*  Wc2T   = (_Float16*) carve((size_t)KCAT * DH * 2);
  _Float16*  WdT    = (_Float16*) carve((size_t)DH * DOUT * 2);
  float*     sum    = (float*)    carve(128 * 4);
  float*     sumsq  = (float*)    carve(128 * 4);
  float*     scale  = (float*)    carve(128 * 4);
  float*     shift  = (float*)    carve(128 * 4);
  float*     colacc = (float*)    carve(DOUT * 4);
  (void)ws_size;
  _Float16* featH = Xh;  // compact [N,DH] f16 staging, aliases Xh when X unused

  const int TB = 256;

  // ---- degrees -> norm ----
  k_fill0<<<ceil_div(N, TB), TB, 0, stream>>>(norm, N);
  k_deg<<<ceil_div(E, TB), TB, 0, stream>>>(dst, norm, E);
  k_norm_inplace<<<ceil_div(N, TB), TB, 0, stream>>>(norm, N);

  // ---- weight (transposed) / feature conversions to f16 ----
  k_cvt_f16<<<ceil_div(NE, TB), TB, 0, stream>>>(features, featH, NE);
  k_cvt_f16_T<<<ceil_div(DH * DH, TB), TB, 0, stream>>>(Wg, WgT, DH, DH);
  k_cvt_f16_T<<<ceil_div(KCAT * DH, TB), TB, 0, stream>>>(Wc1, Wc1T, KCAT, DH);
  k_cvt_f16_T<<<ceil_div(KCAT * DH, TB), TB, 0, stream>>>(Wc2, Wc2T, KCAT, DH);
  k_cvt_f16_T<<<ceil_div(DH * DOUT, TB), TB, 0, stream>>>(Wd, WdT, DH, DOUT);

  // ---- gating: bufH = sigmoid(features @ Wg + bg) ----
  {
    int waves = (N >> 4) * (DH >> 7);
    k_gemm_wmma<1><<<ceil_div(waves, 8), TB, 0, stream>>>(
        featH, DH, WgT, bufH, DH, N, DH, DH, bg, nullptr);
  }

  // ---- two TAGConv + BN + ReLU layers ----
  const _Float16* WcT[2] = {Wc1T, Wc2T};
  const float*    gv[2]  = {g1, g2};
  const float*    bv[2]  = {b1, b2};
  for (int layer = 0; layer < 2; ++layer) {
    // hop 0 -> X[:, 0:128]
    k_copyX<<<ceil_div(NE, TB), TB, 0, stream>>>(bufH, Xh, NE, 0);
    // hop 1
    k_fill0<<<ceil_div(NE, TB), TB, 0, stream>>>(bufAcc, NE);
    k_scatter<<<ceil_div((long)E * 32, TB), TB, 0, stream>>>(
        src, dst, norm, bufH, bufAcc, E * 32);
    k_hopscale<<<ceil_div(NE, TB), TB, 0, stream>>>(bufAcc, norm, bufCur, Xh, NE, DH);
    // hop 2
    k_fill0<<<ceil_div(NE, TB), TB, 0, stream>>>(bufAcc, NE);
    k_scatter<<<ceil_div((long)E * 32, TB), TB, 0, stream>>>(
        src, dst, norm, bufCur, bufAcc, E * 32);
    k_hopscale<<<ceil_div(NE, TB), TB, 0, stream>>>(bufAcc, norm, bufCur, Xh, NE, 2 * DH);
    // concat GEMM: bufAcc = X @ Wc   (K = 384)
    {
      int waves = (N >> 4) * (DH >> 7);
      k_gemm_wmma<0><<<ceil_div(waves, 8), TB, 0, stream>>>(
          Xh, KCAT, WcT[layer], bufAcc, DH, N, KCAT, DH, nullptr, nullptr);
    }
    // BatchNorm (batch stats, biased var) + ReLU -> bufH
    k_fill0<<<1, 256, 0, stream>>>(sum, 256);  // zeros sum(128) + sumsq(128)
    k_bnstats<<<ceil_div(N, 256), 128, 0, stream>>>(bufAcc, sum, sumsq, N);
    k_bnfinal<<<1, 128, 0, stream>>>(sum, sumsq, gv[layer], bv[layer], scale, shift, invN);
    k_bnapply_relu<<<ceil_div(NE, TB), TB, 0, stream>>>(bufAcc, scale, shift, bufH, NE);
  }

  // ---- final GEMM fused with mean readout: out = mean(h @ Wd) + bd ----
  k_cvt_f16<<<ceil_div(NE, TB), TB, 0, stream>>>(bufH, featH, NE);  // X no longer needed
  k_fill0<<<1, DOUT, 0, stream>>>(colacc, DOUT);
  {
    int waves = (N >> 4) * (DOUT >> 7);
    k_gemm_wmma<2><<<ceil_div(waves, 8), TB, 0, stream>>>(
        featH, DH, WdT, nullptr, 0, N, DH, DOUT, nullptr, colacc);
  }
  k_finalout<<<1, DOUT, 0, stream>>>(colacc, bd, out, invN, out_size);
}
